// MultiHeadAttentionBlock_9698036154997
// MI455X (gfx1250) — matrix operand order repaired and checked by reading, not compiled, not run on hardware
//
#include <hip/hip_runtime.h>
#include <hip/hip_bf16.h>
#include <math.h>

typedef __attribute__((ext_vector_type(16))) _Float16 v16h;
typedef __attribute__((ext_vector_type(8)))  float    v8f;

#define D_MODEL   768
#define N_HEADS   12
#define D_K       64
#define S_LEN     4096
#define B_BATCH   2
#define M_ROWS    (B_BATCH * S_LEN)     // 8192
#define NEG_INF_F (-10000000000.0f)

// CDNA5 async global->LDS staging (ASYNCcnt). Set to 0 to fall back to
// plain global_load + ds_store if the assembler rejects the mnemonic.
#define USE_ASYNC_LDS 1

union Frag {
    v16h v;
    _Float16 h[16];
    unsigned int u[8];
};

__device__ __forceinline__ void async_copy_b128(const _Float16* gsrc, _Float16* lds_dst)
{
#if USE_ASYNC_LDS
    // LDS dest address comes from a VGPR; generic LDS pointer's low 32 bits are the
    // LDS byte address (ISA 10.2).
    unsigned ldst = (unsigned)(unsigned long long)lds_dst;
    unsigned long long ga = (unsigned long long)gsrc;
    asm volatile("global_load_async_to_lds_b128 %0, %1, off"
                 :: "v"(ldst), "v"(ga) : "memory");
#else
    uint4 d = *reinterpret_cast<const uint4*>(gsrc);
    *reinterpret_cast<uint4*>(lds_dst) = d;
#endif
}

// wait until at most N async ops remain outstanding (immediate forms)
__device__ __forceinline__ void async_wait_le0()
{
#if USE_ASYNC_LDS
    asm volatile("s_wait_asynccnt 0x0" ::: "memory");
#endif
}
__device__ __forceinline__ void async_wait_le1()
{
#if USE_ASYNC_LDS
    asm volatile("s_wait_asynccnt 0x1" ::: "memory");
#endif
}
__device__ __forceinline__ void async_wait_le2()
{
#if USE_ASYNC_LDS
    asm volatile("s_wait_asynccnt 0x2" ::: "memory");
#endif
}

// ---------------------------------------------------------------------------
// Weight transpose + f32 -> f16 convert:  Wt[n][k] = (f16) W[k][n]
// ---------------------------------------------------------------------------
__global__ __launch_bounds__(256) void transpose_w_f16(
    const float* __restrict__ W, _Float16* __restrict__ Wt)
{
    int idx = blockIdx.x * 256 + threadIdx.x;        // 0 .. 768*768-1
    int kk = idx / D_MODEL;
    int nn = idx % D_MODEL;
    Wt[(size_t)nn * D_MODEL + kk] = (_Float16)W[idx];
}

// ---------------------------------------------------------------------------
// GEMM: C(M=8192 x N=768) = A(f32, row-major, ld=768) @ Wt^T + bias, K=768
//   Block tile: 128(M) x 64(N); 8 waves x (16M x 64N) WMMA tiles.
//   Weight tile (64 n-rows x 32 k, f16, 4KB) double-buffered in LDS via
//   async global->LDS DMA; transfer of step k+1 overlaps WMMA of step k.
// MODE 0: f32 row-major out (final projection -> d_out)
// MODE 1: f16 head-major  out[((b*H+h)*S + s)*64 + d]   (Q, K)
// MODE 2: f16 head-transposed out[((b*H+h)*64 + d)*S + s]  (V -> V^T)
// ---------------------------------------------------------------------------
template <int MODE>
__global__ __launch_bounds__(256) void gemm768_wmma(
    const float* __restrict__ A, const _Float16* __restrict__ Wt,
    const float* __restrict__ bias, void* __restrict__ outp, float out_scale)
{
    __shared__ _Float16 ldsW[2][64 * 32];            // [buf][n_local][k_local]

    const int lane = threadIdx.x & 31;
    const int wave = threadIdx.x >> 5;
    const int ln   = lane & 15;
    const int h2   = lane >> 4;

    const int mblk   = blockIdx.x / 12;              // 64 M blocks of 128
    const int nchunk = blockIdx.x % 12;              // 12 N chunks of 64
    const int mbase  = mblk * 128 + wave * 16;
    const int nbase  = nchunk * 64;

    // staging: thread t copies 8 halves of row (t>>2), chunk (t&3)
    const int wrow  = threadIdx.x >> 2;              // 0..63 (n local)
    const int wpart = threadIdx.x & 3;               // 0..3
    const int wofs  = wrow * 32 + wpart * 8;
    const _Float16* wsrc = Wt + (size_t)(nbase + wrow) * D_MODEL + wpart * 8;

    v8f acc[4] = {v8f{}, v8f{}, v8f{}, v8f{}};
    const float* arow = A + (size_t)(mbase + ln) * D_MODEL;

    const int NIT = D_MODEL / 32;                    // 24 k-steps
    async_copy_b128(wsrc, &ldsW[0][wofs]);           // prologue: tile 0

    for (int it = 0; it < NIT; ++it) {
        const int kb = it * 32;
        const bool more = (it + 1 < NIT);
        if (more) async_copy_b128(wsrc + kb + 32, &ldsW[(it + 1) & 1][wofs]);
        if (more) async_wait_le1(); else async_wait_le0();
        __syncthreads();                             // current tile ready everywhere

        // A fragment (global f32 -> f16): 4 x b128; group g covers elems 4g..4g+3,
        // K start = 4g + 16*h2  (A-layout: K = e + 16h, contiguous)
        Frag af;
        #pragma unroll
        for (int g = 0; g < 4; ++g) {
            int k0 = 4 * g + 16 * h2 + kb;
            float4 f = *reinterpret_cast<const float4*>(arow + k0);
            af.h[4 * g]     = (_Float16)f.x;
            af.h[4 * g + 1] = (_Float16)f.y;
            af.h[4 * g + 2] = (_Float16)f.z;
            af.h[4 * g + 3] = (_Float16)f.w;
        }

        const unsigned int* ldsW_u =
            reinterpret_cast<const unsigned int*>(ldsW[it & 1]);
        #pragma unroll
        for (int nt = 0; nt < 4; ++nt) {
            // B fragment (LDS): lane=col n, elem e -> K = e + 16*h2 (contiguous)
            const int base = ((nt * 16 + ln) * 32 + 16 * h2) >> 1;
            Frag bf;
            #pragma unroll
            for (int p = 0; p < 8; ++p) bf.u[p] = ldsW_u[base + p];
            acc[nt] = __builtin_amdgcn_wmma_f32_16x16x32_f16(
                false, af.v, false, bf.v, (short)0, acc[nt], false, false);
        }
        __syncthreads();                             // done reading before overwrite
    }

    #pragma unroll
    for (int nt = 0; nt < 4; ++nt) {
        const int n  = nbase + nt * 16 + ln;
        const float bv = bias[n];
        #pragma unroll
        for (int r = 0; r < 8; ++r) {
            const int gm  = mbase + r + 8 * h2;      // C: M = r + 8h
            const float val = (acc[nt][r] + bv) * out_scale;
            if (MODE == 0) {
                ((float*)outp)[(size_t)gm * D_MODEL + n] = val;
            } else {
                const int b = gm >> 12, s = gm & (S_LEN - 1);
                const int head = n >> 6, d = n & (D_K - 1);
                const _Float16 hv = (_Float16)val;
                if (MODE == 1)
                    ((_Float16*)outp)[((size_t)(b * N_HEADS + head) * S_LEN + s) * D_K + d] = hv;
                else
                    ((_Float16*)outp)[((size_t)(b * N_HEADS + head) * D_K + d) * S_LEN + s] = hv;
            }
        }
    }
}

// ---------------------------------------------------------------------------
// Attention (transposed flash): block = one (b,head) x 128 queries
// (8 waves x 16-query tiles). Per 32-kv block, the K tile (32x64 f16) and
// V^T tile (64x32 f16) are staged once in LDS via async DMA, double-buffered
// so the transfer of block i+1 overlaps the WMMA/softmax of block i.
//   S^T(kv x q) = K(kv x 64) @ Q^T(64 x q)      -> 4 WMMA / block
//   O^T(d x q) += V^T(d x 32kv) @ P^T(32kv x q) -> 4 WMMA / block
// Online softmax over kv; X output f32 row-major. Q pre-scaled by 1/8.
// ---------------------------------------------------------------------------
__global__ __launch_bounds__(256) void attn_wmma(
    const _Float16* __restrict__ Qh, const _Float16* __restrict__ Kh,
    const _Float16* __restrict__ Vt, const int* __restrict__ mask,
    float* __restrict__ X)
{
    __shared__ _Float16 ldsK[2][32 * 64];            // [buf][kv_local][d]
    __shared__ _Float16 ldsV[2][64 * 32];            // [buf][d][kv_local]

    const int lane = threadIdx.x & 31;
    const int wave = threadIdx.x >> 5;
    const int ln   = lane & 15;
    const int h2   = lane >> 4;

    const int bh   = blockIdx.y;                     // 0 .. 23
    const int b    = bh / N_HEADS;
    const int head = bh - b * N_HEADS;
    const int qbase = (blockIdx.x * 8 + wave) * 16;  // this wave's 16-query tile

    const _Float16* Qb = Qh + (size_t)bh * S_LEN * D_K;
    const _Float16* Kb = Kh + (size_t)bh * S_LEN * D_K;
    const _Float16* Vb = Vt + (size_t)bh * D_K * S_LEN;
    const int* mrow = mask + b * S_LEN;

    // staging assignment
    const int krow  = threadIdx.x >> 3;              // 0..31 (kv local)
    const int kpart = threadIdx.x & 7;               // 8 chunks of 8 halves
    const int vrow  = threadIdx.x >> 2;              // 0..63 (d)
    const int vpart = threadIdx.x & 3;               // 4 chunks of 8 halves
    const int kofs  = krow * 64 + kpart * 8;
    const int vofs  = vrow * 32 + vpart * 8;
    const _Float16* gK0 = Kb + (size_t)krow * D_K + kpart * 8;
    const _Float16* gV0 = Vb + (size_t)vrow * S_LEN + vpart * 8;

    // Q^T B-fragments (lane = query column; reused across whole kv loop)
    Frag qf[2];
    {
        const unsigned int* qp = reinterpret_cast<const unsigned int*>(
            Qb + (size_t)(qbase + ln) * D_K);
        #pragma unroll
        for (int ks = 0; ks < 2; ++ks)
            #pragma unroll
            for (int p = 0; p < 8; ++p)
                qf[ks].u[p] = qp[16 * ks + 8 * h2 + p];
    }

    v8f o[4] = {v8f{}, v8f{}, v8f{}, v8f{}};
    float m_run = -3.0e38f;
    float l_run = 0.0f;

    // prologue: stage kv block 0 into buffer 0
    async_copy_b128(gK0, &ldsK[0][kofs]);
    async_copy_b128(gV0, &ldsV[0][vofs]);

    for (int it = 0; it < S_LEN / 32; ++it) {
        const int kv = it * 32;
        const bool more = (it + 1 < S_LEN / 32);
        if (more) {
            async_copy_b128(gK0 + (size_t)(kv + 32) * D_K, &ldsK[(it + 1) & 1][kofs]);
            async_copy_b128(gV0 + kv + 32,                 &ldsV[(it + 1) & 1][vofs]);
            async_wait_le2();
        } else {
            async_wait_le0();
        }
        const unsigned bal = (unsigned)__ballot(mrow[kv + lane] == 0);
        __syncthreads();                             // current tiles ready

        const unsigned int* ldsK_u =
            reinterpret_cast<const unsigned int*>(ldsK[it & 1]);
        const unsigned int* ldsV_u =
            reinterpret_cast<const unsigned int*>(ldsV[it & 1]);

        // S^T tiles: c0 = kv[0..15] x q16, c1 = kv[16..31] x q16
        v8f c0 = {}, c1 = {};
        #pragma unroll
        for (int tile = 0; tile < 2; ++tile) {
            const int rb = (16 * tile + ln) * 64;    // K-tile row base (halves)
            #pragma unroll
            for (int ks = 0; ks < 2; ++ks) {
                Frag kf;                             // A operand: lane = kv row
                #pragma unroll
                for (int p = 0; p < 8; ++p) {
                    int k0 = 2 * p + 16 * h2 + 32 * ks;
                    kf.u[p] = ldsK_u[(rb + k0) >> 1];
                }
                v8f& cc = tile ? c1 : c0;
                cc = __builtin_amdgcn_wmma_f32_16x16x32_f16(
                    false, kf.v, false, qf[ks].v, (short)0, cc, false, false);
            }
        }

        // mask per key position (32-bit tests): C row M = r + 8*h2 (+16 for c1)
        if (bal) {
            const unsigned mlo = bal >> (8 * h2);
            const unsigned mhi = bal >> (16 + 8 * h2);
            #pragma unroll
            for (int r = 0; r < 8; ++r) {
                if ((mlo >> r) & 1u) c0[r] = NEG_INF_F;
                if ((mhi >> r) & 1u) c1[r] = NEG_INF_F;
            }
        }

        // online softmax (per query column = per lane; halves paired via xor-16)
        float bm = c0[0];
        #pragma unroll
        for (int r = 1; r < 8; ++r) bm = fmaxf(bm, c0[r]);
        #pragma unroll
        for (int r = 0; r < 8; ++r) bm = fmaxf(bm, c1[r]);
        bm = fmaxf(bm, __shfl_xor(bm, 16, 32));
        const float m_new = fmaxf(m_run, bm);
        const float alpha = __expf(m_run - m_new);

        float e0[8], e1[8], ps = 0.0f;
        #pragma unroll
        for (int r = 0; r < 8; ++r) {
            e0[r] = __expf(c0[r] - m_new); ps += e0[r];
            e1[r] = __expf(c1[r] - m_new); ps += e1[r];
        }
        ps += __shfl_xor(ps, 16, 32);
        l_run = l_run * alpha + ps;
        m_run = m_new;
        #pragma unroll
        for (int t = 0; t < 4; ++t) o[t] = o[t] * alpha;

        // P^T B-fragment: element e -> kv = e + 16*h2.
        // Pack (e0,e1) as 2xf16 per dword, one xor-16 shuffle per pair:
        //   h2=0: h[r] = own.e0, h[8+r] = partner.e0
        //   h2=1: h[r] = partner.e1, h[8+r] = own.e1
        Frag pf;
        #pragma unroll
        for (int r = 0; r < 8; ++r) {
            union { unsigned u; _Float16 hh[2]; } own, oth;
            own.hh[0] = (_Float16)e0[r];
            own.hh[1] = (_Float16)e1[r];
            oth.u = (unsigned)__shfl_xor((int)own.u, 16, 32);
            pf.h[r]     = h2 ? oth.hh[1] : own.hh[0];
            pf.h[8 + r] = h2 ? own.hh[1] : oth.hh[0];
        }

        // O^T += V^T @ P^T : 4 d-tiles of 16, contraction K = 32 kv
        #pragma unroll
        for (int t = 0; t < 4; ++t) {
            const int rb = (16 * t + ln) * 32;       // V^T row base (halves)
            Frag vf;                                 // A operand: lane = d row
            #pragma unroll
            for (int p = 0; p < 8; ++p) {
                int k0 = 2 * p + 16 * h2;
                vf.u[p] = ldsV_u[(rb + k0) >> 1];
            }
            o[t] = __builtin_amdgcn_wmma_f32_16x16x32_f16(
                false, vf.v, false, pf.v, (short)0, o[t], false, false);
        }
        __syncthreads();                             // done reading before overwrite
    }

    // normalize and write X[b, s, head*64 + d]  (lane = query s, rows = d)
    const float inv_l = 1.0f / l_run;
    const int s = qbase + ln;
    float* xrow = X + (size_t)(b * S_LEN + s) * D_MODEL + head * D_K;
    #pragma unroll
    for (int t = 0; t < 4; ++t)
        #pragma unroll
        for (int r = 0; r < 8; ++r)
            xrow[16 * t + r + 8 * h2] = o[t][r] * inv_l;
}

// ---------------------------------------------------------------------------
extern "C" void kernel_launch(void* const* d_in, const int* in_sizes, int n_in,
                              void* d_out, int out_size, void* d_ws, size_t ws_size,
                              hipStream_t stream)
{
    const float* q    = (const float*)d_in[0];
    const float* k    = (const float*)d_in[1];
    const float* v    = (const float*)d_in[2];
    const int*   mask = (const int*)  d_in[3];
    const float* Wq   = (const float*)d_in[4];
    const float* bq   = (const float*)d_in[5];
    const float* Wk   = (const float*)d_in[6];
    const float* bk   = (const float*)d_in[7];
    const float* Wv   = (const float*)d_in[8];
    const float* bv   = (const float*)d_in[9];
    const float* Wo   = (const float*)d_in[10];
    const float* bo   = (const float*)d_in[11];

    char* ws = (char*)d_ws;
    size_t off = 0;
    auto salloc = [&](size_t bytes) -> void* {
        void* p = ws + off;
        off += (bytes + 255) & ~(size_t)255;
        return p;
    };
    const size_t WT_BYTES  = (size_t)D_MODEL * D_MODEL * sizeof(_Float16);
    const size_t QKV_BYTES = (size_t)M_ROWS * D_MODEL * sizeof(_Float16);
    _Float16* WtQ = (_Float16*)salloc(WT_BYTES);
    _Float16* WtK = (_Float16*)salloc(WT_BYTES);
    _Float16* WtV = (_Float16*)salloc(WT_BYTES);
    _Float16* WtO = (_Float16*)salloc(WT_BYTES);
    _Float16* Qh  = (_Float16*)salloc(QKV_BYTES);
    _Float16* Kh  = (_Float16*)salloc(QKV_BYTES);
    _Float16* Vt  = (_Float16*)salloc(QKV_BYTES);
    float*    X   = (float*)   salloc((size_t)M_ROWS * D_MODEL * sizeof(float));

    const dim3 blk(256);
    const dim3 tgrid((D_MODEL * D_MODEL) / 256);
    transpose_w_f16<<<tgrid, blk, 0, stream>>>(Wq, WtQ);
    transpose_w_f16<<<tgrid, blk, 0, stream>>>(Wk, WtK);
    transpose_w_f16<<<tgrid, blk, 0, stream>>>(Wv, WtV);
    transpose_w_f16<<<tgrid, blk, 0, stream>>>(Wo, WtO);

    // 8192x768: block tile 128M x 64N -> 64*12 = 768 blocks
    const dim3 ggrid(768);
    const float qscale = 0.125f;   // 1/sqrt(D_K) folded into Q projection
    gemm768_wmma<1><<<ggrid, blk, 0, stream>>>(q, WtQ, bq, Qh, qscale);
    gemm768_wmma<1><<<ggrid, blk, 0, stream>>>(k, WtK, bk, Kh, 1.0f);
    gemm768_wmma<2><<<ggrid, blk, 0, stream>>>(v, WtV, bv, Vt, 1.0f);

    attn_wmma<<<dim3(S_LEN / (16 * 8), B_BATCH * N_HEADS), blk, 0, stream>>>(
        Qh, Kh, Vt, mask, X);

    gemm768_wmma<0><<<ggrid, blk, 0, stream>>>(X, WtO, bo, d_out, 1.0f);
}
